// Mamba_21431886807210
// MI455X (gfx1250) — matrix operand order repaired and checked by reading, not compiled, not run on hardware
//
#include <hip/hip_runtime.h>
#include <hip/hip_bf16.h>

// ---------------------------------------------------------------------------
// Bidirectional Mamba block for MI455X (gfx1250, wave32, WMMA).
// Big GEMMs (in_proj 77 GF, out_proj 39 GF) use v_wmma_f32_16x16x32_bf16 with
// async global->LDS double-buffered B tiles and 32x64 wave tiles.
// ---------------------------------------------------------------------------

typedef __attribute__((ext_vector_type(16))) __bf16 v16bf;
typedef __attribute__((ext_vector_type(8)))  float  v8f;
typedef __attribute__((ext_vector_type(4)))  int    v4i;

#define B_SZ    64
#define LSEQ    256
#define DMODEL  768
#define DINNER  1536
#define DSTATE  16
#define DCONV   4
#define DTRANK  48
#define NROWS   16

#if defined(__has_builtin)
#  if __has_builtin(__builtin_amdgcn_global_load_async_to_lds_b128)
#    define USE_ASYNC_LDS 1
#  endif
#endif
#ifndef USE_ASYNC_LDS
#  define USE_ASYNC_LDS 0
#endif

__device__ __forceinline__ unsigned short f2bf(float f) {
    unsigned u = __float_as_uint(f);
    unsigned r = u + 0x7FFFu + ((u >> 16) & 1u);
    return (unsigned short)(r >> 16);
}

__device__ __forceinline__ float softplus_f(float x) {
    return (x > 20.f) ? x : log1pf(__expf(x));
}
__device__ __forceinline__ float sigmoid_f(float x) {
    return 1.f / (1.f + __expf(-x));
}

#if USE_ASYNC_LDS
__device__ __forceinline__ void async_copy_b128(const unsigned short* g,
                                                unsigned short* l) {
    __builtin_amdgcn_global_load_async_to_lds_b128(
        (__attribute__((address_space(1))) v4i*)g,
        (__attribute__((address_space(3))) v4i*)l,
        0, 0);
}
#endif

// ---------------------------------------------------------------------------
// f32 -> bf16 (RNE), 4 elements per thread
// ---------------------------------------------------------------------------
__global__ __launch_bounds__(256) void cvt_f32_bf16(const float* __restrict__ src,
                                                    unsigned short* __restrict__ dst,
                                                    int n) {
    int i = (blockIdx.x * 256 + threadIdx.x) * 4;
    if (i >= n) return;
    float4 v = *(const float4*)(src + i);
    uint2 o;
    o.x = (unsigned)f2bf(v.x) | ((unsigned)f2bf(v.y) << 16);
    o.y = (unsigned)f2bf(v.z) | ((unsigned)f2bf(v.w) << 16);
    *(uint2*)(dst + i) = o;
}

// ---------------------------------------------------------------------------
// bf16 WMMA GEMM: C(M x N) = A(M x K, row-major) * Bw(N x K, row-major)^T
//   LAYOUT 0: C row-major [m*N + n]                  (out_proj -> d_out)
//   LAYOUT 1: C as (b, n, l) with m = b*256 + l      (in_proj  -> xz)
// Block: 256 threads = 8 waves. Block tile 256(M) x 64(N).
// Wave tile 32 x 64: 8 accumulators, 2 A-frags; each B-frag feeds 2 WMMAs.
// B tile (64x32 bf16) double-buffered in LDS via async global->LDS copies;
// one barrier per K-step.
// ---------------------------------------------------------------------------
union FragU { uint4 u[2]; v16bf v; };

template <int LAYOUT>
__global__ __launch_bounds__(256) void wmma_gemm_bf16(
    const unsigned short* __restrict__ A,
    const unsigned short* __restrict__ Bw,
    float* __restrict__ C,
    int M, int N, int K) {

    __shared__ unsigned short sB[2][64][40];   // double-buffered B tile (+pad)
    __shared__ float sC[8][16 * 36];           // per-wave frag transpose staging

    const int tid  = threadIdx.x;
    const int wave = tid >> 5;
    const int lane = tid & 31;
    const int half = lane >> 4;       // 0: lanes 0-15, 1: lanes 16-31
    const int l16  = lane & 15;

    const int n0 = blockIdx.x * 64;
    const int m0 = blockIdx.y * 256 + wave * 32;

    v8f acc0[4], acc1[4];
    #pragma unroll
    for (int f = 0; f < 4; ++f)
        #pragma unroll
        for (int j = 0; j < 8; ++j) { acc0[f][j] = 0.f; acc1[f][j] = 0.f; }

    const int brow = tid >> 2;            // 0..63
    const int bkc  = (tid & 3) * 8;       // 0,8,16,24
    const unsigned short* bsrc = &Bw[(size_t)(n0 + brow) * K + bkc];

    // prologue: fetch tile 0
#if USE_ASYNC_LDS
    async_copy_b128(bsrc, &sB[0][brow][bkc]);
    asm volatile("s_wait_asynccnt 0x0" ::: "memory");
#else
    *(uint4*)&sB[0][brow][bkc] = *(const uint4*)bsrc;
#endif
    __syncthreads();

    int buf = 0;
    for (int k0 = 0; k0 < K; k0 += 32) {
        const bool has_next = (k0 + 32 < K);
#if USE_ASYNC_LDS
        if (has_next) async_copy_b128(bsrc + k0 + 32, &sB[buf ^ 1][brow][bkc]);
#else
        uint4 nxt;
        if (has_next) nxt = *(const uint4*)(bsrc + k0 + 32);
#endif

        // A fragments (CDNA5 16-bit A 16x32 striping): lane holds one row;
        // elems 0..7 = K[8h..], elems 8..15 = K[16+8h..]
        FragU a0, a1;
        const size_t ab0 = (size_t)(m0 + l16) * K + k0;
        const size_t ab1 = (size_t)(m0 + 16 + l16) * K + k0;
        a0.u[0] = *(const uint4*)&A[ab0 + half * 8];
        a0.u[1] = *(const uint4*)&A[ab0 + 16 + half * 8];
        a1.u[0] = *(const uint4*)&A[ab1 + half * 8];
        a1.u[1] = *(const uint4*)&A[ab1 + 16 + half * 8];

        #pragma unroll
        for (int f = 0; f < 4; ++f) {
            FragU bfr;
            const int nl = f * 16 + l16;
            bfr.u[0] = *(const uint4*)&sB[buf][nl][half * 8];
            bfr.u[1] = *(const uint4*)&sB[buf][nl][16 + half * 8];
            acc0[f] = __builtin_amdgcn_wmma_f32_16x16x32_bf16(
                false, a0.v, false, bfr.v, (short)0, acc0[f], false, false);
            acc1[f] = __builtin_amdgcn_wmma_f32_16x16x32_bf16(
                false, a1.v, false, bfr.v, (short)0, acc1[f], false, false);
        }

#if USE_ASYNC_LDS
        if (has_next) asm volatile("s_wait_asynccnt 0x0" ::: "memory");
#else
        if (has_next) *(uint4*)&sB[buf ^ 1][brow][bkc] = nxt;
#endif
        __syncthreads();
        buf ^= 1;
    }

    if (LAYOUT == 0) {
        // row-major: per register r, lanes 0-15 cover a contiguous n-run
        #pragma unroll
        for (int f = 0; f < 4; ++f) {
            const int n = n0 + f * 16 + l16;
            #pragma unroll
            for (int r = 0; r < 8; ++r) {
                C[(size_t)(m0 + r + 8 * half) * N + n]      = acc0[f][r];
                C[(size_t)(m0 + 16 + r + 8 * half) * N + n] = acc1[f][r];
            }
        }
    } else {
        // (b, n, l): transpose one 16x32 fragment at a time through LDS
        // (wave-local; LDS ops are in-order per wave) -> float4 stores along l
        float* sc = &sC[wave][0];
        const int b  = blockIdx.y;          // GTILE_M == LSEQ == 256
        const int l0 = wave * 32;
        const int m4 = (lane & 7) * 4;
        #pragma unroll
        for (int f = 0; f < 4; ++f) {
            #pragma unroll
            for (int r = 0; r < 8; ++r) {
                sc[l16 * 36 + r + 8 * half]      = acc0[f][r];
                sc[l16 * 36 + 16 + r + 8 * half] = acc1[f][r];
            }
            #pragma unroll
            for (int pass = 0; pass < 4; ++pass) {
                const int nl = pass * 4 + (lane >> 3);
                float4 v = *(float4*)&sc[nl * 36 + m4];
                *(float4*)&C[((size_t)b * N + (n0 + f * 16 + nl)) * LSEQ + l0 + m4] = v;
            }
        }
    }
}

// ---------------------------------------------------------------------------
// depthwise causal conv (K=4) + SiLU + 16-wide mean pool.
// One wave per (b, d) row of 256 elements; `reverse` flips the sequence.
// ---------------------------------------------------------------------------
__global__ __launch_bounds__(256) void conv_silu_pool(
    const float* __restrict__ xz,          // (B, 2*DINNER, L); x half used
    const float* __restrict__ w,           // (DINNER, 4)
    const float* __restrict__ bias,        // (DINNER,)
    float* __restrict__ xc,                // (B, DINNER, L)
    float* __restrict__ xcomp,             // (B, DINNER, 16)
    int reverse) {

    __shared__ float srow[8][260];
    const int wave = threadIdx.x >> 5;
    const int lane = threadIdx.x & 31;
    const int row  = blockIdx.x * 8 + wave;       // (b, d) flattened
    const int b = row / DINNER;
    const int d = row % DINNER;

    const float* src = &xz[((size_t)b * (2 * DINNER) + d) * LSEQ];
    if (lane < 3) srow[wave][lane] = 0.f;
    #pragma unroll
    for (int i = 0; i < 8; ++i) {
        const int l = i * 32 + lane;
        srow[wave][3 + l] = reverse ? src[LSEQ - 1 - l] : src[l];
    }
    __syncthreads();

    const float w0 = w[d * 4 + 0], w1 = w[d * 4 + 1];
    const float w2 = w[d * 4 + 2], w3 = w[d * 4 + 3];
    const float bb = bias[d];
    float yv[8];
    #pragma unroll
    for (int i = 0; i < 8; ++i) {
        const int l = i * 32 + lane;
        float a = bb + w0 * srow[wave][l] + w1 * srow[wave][l + 1]
                     + w2 * srow[wave][l + 2] + w3 * srow[wave][l + 3];
        yv[i] = a * sigmoid_f(a);
        xc[(size_t)row * LSEQ + l] = yv[i];
    }
    __syncthreads();
    #pragma unroll
    for (int i = 0; i < 8; ++i) srow[wave][3 + i * 32 + lane] = yv[i];
    __syncthreads();
    if (lane < 16) {
        float s = 0.f;
        #pragma unroll
        for (int j = 0; j < 16; ++j) s += srow[wave][3 + lane * 16 + j];
        xcomp[(size_t)row * NROWS + lane] = s * (1.f / 16.f);
    }
}

// ---------------------------------------------------------------------------
// x_dbl[b, row, e] = sum_d xcomp[b, d, row] * x_proj_w[e, d]   (e < 80)
// ---------------------------------------------------------------------------
__global__ __launch_bounds__(256) void xproj_kernel(
    const float* __restrict__ xcomp, const float* __restrict__ xpw,
    float* __restrict__ xdbl) {
    const int idx = blockIdx.x * 256 + threadIdx.x;   // 64*16*80
    const int e   = idx % 80;
    const int row = (idx / 80) % NROWS;
    const int b   = idx / (80 * NROWS);
    float s = 0.f;
    const float* xp = &xcomp[(size_t)b * DINNER * NROWS + row];
    const float* wp = &xpw[(size_t)e * DINNER];
    for (int d = 0; d < DINNER; ++d) s += xp[d * NROWS] * wp[d];
    xdbl[idx] = s;
}

// ---------------------------------------------------------------------------
// dt[b, d, row] = sum_r xdbl[b, row, r] * dt_proj_w[d, r]   (r < 48)
// ---------------------------------------------------------------------------
__global__ __launch_bounds__(256) void dtproj_kernel(
    const float* __restrict__ xdbl, const float* __restrict__ dtw,
    float* __restrict__ dt) {
    const int idx = blockIdx.x * 256 + threadIdx.x;   // 64*1536*16
    const int row = idx % NROWS;
    const int d   = (idx / NROWS) % DINNER;
    const int b   = idx / (NROWS * DINNER);
    const float* xp = &xdbl[((size_t)b * NROWS + row) * 80];
    const float* wp = &dtw[(size_t)d * DTRANK];
    float s = 0.f;
    #pragma unroll 8
    for (int r = 0; r < DTRANK; ++r) s += xp[r] * wp[r];
    dt[idx] = s;
}

// ---------------------------------------------------------------------------
// selective scan: one thread per (b, d); 16 steps, N=16 state in registers
// ---------------------------------------------------------------------------
__global__ __launch_bounds__(256) void scan_kernel(
    const float* __restrict__ xcomp, const float* __restrict__ dt,
    const float* __restrict__ xdbl,  const float* __restrict__ A_log,
    const float* __restrict__ dt_bias, float* __restrict__ y) {
    const int idx = blockIdx.x * 256 + threadIdx.x;   // 64*1536
    const int d = idx % DINNER;
    const int b = idx / DINNER;

    float Acoef[DSTATE];
    #pragma unroll
    for (int n = 0; n < DSTATE; ++n) Acoef[n] = -__expf(A_log[d * DSTATE + n]);
    float h[DSTATE];
    #pragma unroll
    for (int n = 0; n < DSTATE; ++n) h[n] = 0.f;

    const float  db  = dt_bias[d];
    const float* dtp = &dt[(size_t)idx * NROWS];
    const float* xcp = &xcomp[(size_t)idx * NROWS];
    const float* xd  = &xdbl[(size_t)b * NROWS * 80];
    float*       yp  = &y[(size_t)idx * NROWS];

    for (int row = 0; row < NROWS; ++row) {
        const float delta = softplus_f(dtp[row] + db);
        const float du    = delta * xcp[row];
        const float* bc   = &xd[row * 80 + DTRANK];
        float yv = 0.f;
        #pragma unroll
        for (int n = 0; n < DSTATE; ++n) {
            h[n] = __expf(delta * Acoef[n]) * h[n] + du * bc[n];
            yv += h[n] * bc[DSTATE + n];
        }
        yp[row] = yv;
    }
}

// ---------------------------------------------------------------------------
// combine fwd/bwd + LayerNorm(1536) + SiLU(z) gate -> bf16 activations
// One block per (b, l); each thread owns 6 channels.
// ---------------------------------------------------------------------------
__global__ __launch_bounds__(256) void comb_ln_gate(
    const float* __restrict__ yf, const float* __restrict__ yb,
    const float* __restrict__ xcf, const float* __restrict__ xcb,
    const float* __restrict__ Dp,  const float* __restrict__ Dpb,
    const float* __restrict__ xz,
    const float* __restrict__ gamma, const float* __restrict__ beta,
    unsigned short* __restrict__ gated) {

    __shared__ float sred[512];
    const int t  = threadIdx.x;
    const int b  = blockIdx.x >> 8;
    const int l  = blockIdx.x & 255;
    const int lb = LSEQ - 1 - l;

    float v[6];
    float sum = 0.f, sumsq = 0.f;
    #pragma unroll
    for (int i = 0; i < 6; ++i) {
        const int d = t + i * 256;
        const size_t rd = (size_t)b * DINNER + d;
        const float a = yf[rd * NROWS + (l >> 4)]  + Dp[d]  * xcf[rd * LSEQ + l];
        const float c = yb[rd * NROWS + (lb >> 4)] + Dpb[d] * xcb[rd * LSEQ + lb];
        v[i] = 0.5f * (a + c);
        sum += v[i];
        sumsq += v[i] * v[i];
    }
    sred[t] = sum; sred[256 + t] = sumsq;
    __syncthreads();
    for (int s = 128; s > 0; s >>= 1) {
        if (t < s) { sred[t] += sred[t + s]; sred[256 + t] += sred[256 + t + s]; }
        __syncthreads();
    }
    const float mean = sred[0] * (1.f / DINNER);
    const float var  = sred[256] * (1.f / DINNER) - mean * mean;
    const float inv  = rsqrtf(var + 1e-5f);

    #pragma unroll
    for (int i = 0; i < 6; ++i) {
        const int d = t + i * 256;
        const float nrm = (v[i] - mean) * inv * gamma[d] + beta[d];
        const float z = xz[((size_t)b * (2 * DINNER) + DINNER + d) * LSEQ + l];
        const float g = nrm * z * sigmoid_f(z);
        gated[((size_t)(b * LSEQ + l)) * DINNER + d] = f2bf(g);
    }
}

// ---------------------------------------------------------------------------
// launch
// ---------------------------------------------------------------------------
extern "C" void kernel_launch(void* const* d_in, const int* in_sizes, int n_in,
                              void* d_out, int out_size, void* d_ws, size_t ws_size,
                              hipStream_t stream) {
    const float* hs      = (const float*)d_in[0];
    const float* w_in    = (const float*)d_in[1];
    const float* conv_w  = (const float*)d_in[2];
    const float* conv_b  = (const float*)d_in[3];
    const float* conv_wb = (const float*)d_in[4];
    const float* conv_bb = (const float*)d_in[5];
    const float* xpw     = (const float*)d_in[6];
    const float* xpw_b   = (const float*)d_in[7];
    const float* dtw     = (const float*)d_in[8];
    const float* dtw_b   = (const float*)d_in[9];
    const float* dtb     = (const float*)d_in[10];
    const float* dtb_b   = (const float*)d_in[11];
    const float* A_log   = (const float*)d_in[12];
    const float* Ab_log  = (const float*)d_in[13];
    const float* Dp      = (const float*)d_in[14];
    const float* Dp_b    = (const float*)d_in[15];
    const float* ln_g    = (const float*)d_in[16];
    const float* ln_b    = (const float*)d_in[17];
    const float* w_out   = (const float*)d_in[18];
    float* out = (float*)d_out;

    // workspace carve-out (256B aligned)
    char* base = (char*)d_ws;
    size_t off = 0;
    auto carve = [&](size_t bytes) {
        char* p = base + off;
        off += (bytes + 255) & ~(size_t)255;
        return p;
    };
    const size_t M1 = (size_t)B_SZ * LSEQ;                         // 16384
    unsigned short* hs_bf    = (unsigned short*)carve(M1 * DMODEL * 2);
    unsigned short* win_bf   = (unsigned short*)carve((size_t)2 * DINNER * DMODEL * 2);
    unsigned short* wout_bf  = (unsigned short*)carve((size_t)DMODEL * DINNER * 2);
    float*          xz       = (float*)carve((size_t)B_SZ * 2 * DINNER * LSEQ * 4);
    float*          xc_f     = (float*)carve((size_t)B_SZ * DINNER * LSEQ * 4);
    float*          xc_b     = (float*)carve((size_t)B_SZ * DINNER * LSEQ * 4);
    float*          xcomp_f  = (float*)carve((size_t)B_SZ * DINNER * NROWS * 4);
    float*          xcomp_b  = (float*)carve((size_t)B_SZ * DINNER * NROWS * 4);
    float*          xdbl_f   = (float*)carve((size_t)B_SZ * NROWS * 80 * 4);
    float*          xdbl_b   = (float*)carve((size_t)B_SZ * NROWS * 80 * 4);
    float*          dt_f     = (float*)carve((size_t)B_SZ * DINNER * NROWS * 4);
    float*          dt_b     = (float*)carve((size_t)B_SZ * DINNER * NROWS * 4);
    float*          y_f      = (float*)carve((size_t)B_SZ * DINNER * NROWS * 4);
    float*          y_b      = (float*)carve((size_t)B_SZ * DINNER * NROWS * 4);
    unsigned short* gated_bf = (unsigned short*)carve(M1 * DINNER * 2);

    // 1) bf16 conversions
    {
        int n0 = (int)(M1 * DMODEL);
        cvt_f32_bf16<<<n0 / 1024, 256, 0, stream>>>(hs, hs_bf, n0);
        int n1 = 2 * DINNER * DMODEL;
        cvt_f32_bf16<<<n1 / 1024, 256, 0, stream>>>(w_in, win_bf, n1);
        int n2 = DMODEL * DINNER;
        cvt_f32_bf16<<<n2 / 1024, 256, 0, stream>>>(w_out, wout_bf, n2);
    }

    // 2) in_proj GEMM: xz(b, e, l) = hs(m, d) * w_in(e, d)^T
    {
        dim3 grid((2 * DINNER) / 64, (int)(M1 / 256));
        wmma_gemm_bf16<1><<<grid, 256, 0, stream>>>(hs_bf, win_bf, xz,
                                                    (int)M1, 2 * DINNER, DMODEL);
    }

    // 3) conv + SiLU + pool, both directions
    {
        const int blocks = (B_SZ * DINNER) / 8;
        conv_silu_pool<<<blocks, 256, 0, stream>>>(xz, conv_w,  conv_b,  xc_f, xcomp_f, 0);
        conv_silu_pool<<<blocks, 256, 0, stream>>>(xz, conv_wb, conv_bb, xc_b, xcomp_b, 1);
    }

    // 4) x_proj matvec
    {
        const int total = B_SZ * NROWS * 80;
        xproj_kernel<<<total / 256, 256, 0, stream>>>(xcomp_f, xpw,   xdbl_f);
        xproj_kernel<<<total / 256, 256, 0, stream>>>(xcomp_b, xpw_b, xdbl_b);
    }

    // 5) dt_proj matvec
    {
        const int total = B_SZ * DINNER * NROWS;
        dtproj_kernel<<<total / 256, 256, 0, stream>>>(xdbl_f, dtw,   dt_f);
        dtproj_kernel<<<total / 256, 256, 0, stream>>>(xdbl_b, dtw_b, dt_b);
    }

    // 6) selective scan
    {
        const int total = B_SZ * DINNER;
        scan_kernel<<<total / 256, 256, 0, stream>>>(xcomp_f, dt_f, xdbl_f, A_log,  dtb,   y_f);
        scan_kernel<<<total / 256, 256, 0, stream>>>(xcomp_b, dt_b, xdbl_b, Ab_log, dtb_b, y_b);
    }

    // 7) combine + LN + gate -> bf16
    comb_ln_gate<<<(int)M1, 256, 0, stream>>>(y_f, y_b, xc_f, xc_b, Dp, Dp_b,
                                              xz, ln_g, ln_b, gated_bf);

    // 8) out_proj GEMM: out(m, o) = gated(m, d) * w_out(o, d)^T
    {
        dim3 grid(DMODEL / 64, (int)(M1 / 256));
        wmma_gemm_bf16<0><<<grid, 256, 0, stream>>>(gated_bf, wout_bf, out,
                                                    (int)M1, DMODEL, DINNER);
    }
}
